// pointnet_fp_module_32298154066788
// MI455X (gfx1250) — matrix operand order repaired and checked by reading, not compiled, not run on hardware
//
#include <hip/hip_runtime.h>
#include <hip/hip_bf16.h>

typedef __attribute__((ext_vector_type(16))) _Float16 v16h;
typedef __attribute__((ext_vector_type(8)))  _Float16 v8h;
typedef __attribute__((ext_vector_type(2)))  _Float16 v2h;
typedef __attribute__((ext_vector_type(8)))  float    v8f;

namespace {
constexpr int kB  = 16;
constexpr int kN1 = 4096;
constexpr int kN2 = 1024;
constexpr int kC1 = 128;
constexpr int kC2 = 256;
constexpr int kCX = kC2 + kC1;   // 384 concat channels
constexpr int kO1 = 256;
constexpr int kO2 = 128;

constexpr int kNT  = 64;         // query points per workgroup (fused kernel)
constexpr int kXR  = kCX + 8;    // padded X row (halves): 392 -> stride 196 dwords == 4 mod 64
constexpr int kHR  = kO1 + 8;    // padded H row (halves): 264 -> stride 132 dwords == 4 mod 64
constexpr int kNTT = kNT / 16;   // n-tiles per workgroup (4)
constexpr int kK1  = kCX / 32;   // 12 K-steps, layer 1
constexpr int kK2  = kO1 / 32;   //  8 K-steps, layer 2
constexpr size_t kSmemBytes = (size_t)(kNT * kXR + kNT * kHR) * sizeof(_Float16); // 83,968 B
}

// ---------------------------------------------------------------------------
// Kernel 1: brute-force 3-NN + inverse-distance weights.
// Rank by (|q|^2 - 2 p.q); add |p|^2 + clamp only for the 3 winners.
// ---------------------------------------------------------------------------
__global__ __launch_bounds__(256) void knn3_weights_kernel(
    const float* __restrict__ xyz1, const float* __restrict__ xyz2,
    int* __restrict__ idx3, float* __restrict__ w3)
{
  __shared__ float4 pts[kN2];                       // 16 KB: {x,y,z,|q|^2}
  const int b = blockIdx.y;
  const int n = blockIdx.x * 256 + threadIdx.x;

  const float* q = xyz2 + b * 3 * kN2;
  for (int m = threadIdx.x; m < kN2; m += 256) {
    float qx = q[m], qy = q[kN2 + m], qz = q[2 * kN2 + m];
    pts[m] = make_float4(qx, qy, qz, qx * qx + qy * qy + qz * qz);
  }
  __syncthreads();

  const float* p = xyz1 + b * 3 * kN1;
  const float px = p[n], py = p[kN1 + n], pz = p[2 * kN1 + n];
  const float nx = -2.f * px, ny = -2.f * py, nz = -2.f * pz;

  float d0 = 1e30f, d1 = 1e30f, d2 = 1e30f;
  int   i0 = 0,     i1 = 0,     i2 = 0;
  #pragma unroll 4
  for (int m = 0; m < kN2; ++m) {
    float4 t = pts[m];                               // LDS broadcast read
    float d = fmaf(nx, t.x, fmaf(ny, t.y, fmaf(nz, t.z, t.w)));
    if (d < d2) {
      if (d < d1) {
        if (d < d0) { d2 = d1; i2 = i1; d1 = d0; i1 = i0; d0 = d; i0 = m; }
        else        { d2 = d1; i2 = i1; d1 = d;  i1 = m; }
      } else        { d2 = d;  i2 = m; }
    }
  }
  const float sq1 = px * px + py * py + pz * pz;
  d0 = fmaxf(d0 + sq1, 0.f);
  d1 = fmaxf(d1 + sq1, 0.f);
  d2 = fmaxf(d2 + sq1, 0.f);
  const float r0 = 1.f / (d0 + 1e-8f);
  const float r1 = 1.f / (d1 + 1e-8f);
  const float r2 = 1.f / (d2 + 1e-8f);
  const float inv = 1.f / (r0 + r1 + r2);

  const int base = (b * kN1 + n) * 3;
  idx3[base + 0] = i0; idx3[base + 1] = i1; idx3[base + 2] = i2;
  w3[base + 0] = r0 * inv; w3[base + 1] = r1 * inv; w3[base + 2] = r2 * inv;
}

// ---------------------------------------------------------------------------
// Kernel 2: one-shot f32 -> f16 conversion of the MLP weights.
// ---------------------------------------------------------------------------
__global__ __launch_bounds__(256) void cvt_weights_kernel(
    const float* __restrict__ W1, const float* __restrict__ W2,
    _Float16* __restrict__ W1h, _Float16* __restrict__ W2h)
{
  const int t = blockIdx.x * 256 + threadIdx.x;
  if (t < kO1 * kCX) W1h[t] = (_Float16)W1[t];
  const int u = t - kO1 * kCX;
  if (u >= 0 && u < kO2 * kO1) W2h[u] = (_Float16)W2[u];
}

// ---------------------------------------------------------------------------
// Kernel 3: fused interpolate + concat + MLP1(relu) + MLP2(relu) using
// v_wmma_f32_16x16x32_f16. One workgroup = one batch x 64 points.
// Weight (A) fragments are held in VGPRs per o-tile; activations (B) stream
// from padded LDS. Dynamic LDS (84 KB) exercises the 320 KB/WGP pool.
// ---------------------------------------------------------------------------
__global__ __launch_bounds__(256) void fused_mlp_kernel(
    const float* __restrict__ f1, const float* __restrict__ f2,
    const float* __restrict__ bias1, const float* __restrict__ bias2,
    const _Float16* __restrict__ W1h, const _Float16* __restrict__ W2h,
    const int* __restrict__ idx3, const float* __restrict__ w3,
    float* __restrict__ out)
{
  extern __shared__ _Float16 smem[];
  _Float16* Xs = smem;               // kNT * kXR halves : concat activations, point-major
  _Float16* Hs = smem + kNT * kXR;   // kNT * kHR halves : layer-1 output (f16)

  const int b   = blockIdx.y;
  const int n0  = blockIdx.x * kNT;
  const int tid = threadIdx.x;

  // ---- Phase 1: build X[n][0:384] = [3-pt interp of features2 | features1] ----
  {
    const int nl  = tid & (kNT - 1);       // point within tile; wave spans 32 consecutive n
    const int sub = tid >> 6;              // 0..3 channel slice
    const int ng  = n0 + nl;
    const int b3  = (b * kN1 + ng) * 3;
    const int j0 = idx3[b3 + 0], j1 = idx3[b3 + 1], j2 = idx3[b3 + 2];
    const float w0 = w3[b3 + 0], w1 = w3[b3 + 1], w2 = w3[b3 + 2];

    const float* f2b = f2 + b * kC2 * kN2;
    _Float16* xrow = Xs + nl * kXR;
    #pragma unroll 4
    for (int cc = 0; cc < kC2 / 4; ++cc) {           // 64 interp channels / thread
      const int c = sub * (kC2 / 4) + cc;
      const float* r = f2b + c * kN2;                // 4 KB row: gather stays in-page
      xrow[c] = (_Float16)(w0 * r[j0] + w1 * r[j1] + w2 * r[j2]);
    }
    const float* f1b = f1 + b * kC1 * kN1;
    #pragma unroll 4
    for (int cc = 0; cc < kC1 / 4; ++cc) {           // 32 passthrough channels / thread
      const int c = sub * (kC1 / 4) + cc;
      xrow[kC2 + c] = (_Float16)f1b[c * kN1 + ng];   // coalesced in n
    }
  }
  __syncthreads();

  const int lane = tid & 31;
  const int wave = tid >> 5;           // 0..7
  const int l15  = lane & 15;
  const int hi   = lane >> 4;          // 0 = lanes 0-15, 1 = lanes 16-31
  const int hi8  = hi * 8;

  // ---- Phase 2: layer 1  H = relu(W1 * X + b1); A resident, B streamed ----
  for (int ot = wave; ot < kO1 / 16; ot += 8) {      // 2 o-tiles per wave
    v16h Afr[kK1];                                    // 96 VGPRs of resident weights
    const _Float16* wrl = W1h + (ot * 16 + l15) * kCX + hi8;
    #pragma unroll
    for (int k = 0; k < kK1; ++k) {
      v8h a0 = *(const v8h*)(wrl + 32 * k);
      v8h a1 = *(const v8h*)(wrl + 32 * k + 16);
      Afr[k] = __builtin_shufflevector(a0, a1, 0,1,2,3,4,5,6,7,8,9,10,11,12,13,14,15);
    }
    for (int nt = 0; nt < kNTT; ++nt) {              // 4 n-tiles reuse A
      const int nl = nt * 16 + l15;
      v8f acc;
      #pragma unroll
      for (int r = 0; r < 8; ++r) acc[r] = bias1[ot * 16 + hi8 + r];

      const _Float16* xrow = Xs + nl * kXR + hi * 8;
      #pragma unroll
      for (int k = 0; k < kK1; ++k) {
        v8h c0 = *(const v8h*)(xrow + 32 * k);
        v8h c1 = *(const v8h*)(xrow + 32 * k + 16);
        v16h Bm = __builtin_shufflevector(c0, c1, 0,1,2,3,4,5,6,7,8,9,10,11,12,13,14,15);
        acc = __builtin_amdgcn_wmma_f32_16x16x32_f16(false, Afr[k], false, Bm,
                                                     (short)0, acc, false, false);
      }
      _Float16* hrow = Hs + nl * kHR + ot * 16 + hi8;  // D: M = r + 8*hi
      #pragma unroll
      for (int r = 0; r < 8; r += 2) {                 // packed 32-bit DS stores
        v2h p;
        p[0] = (_Float16)fmaxf(acc[r], 0.f);
        p[1] = (_Float16)fmaxf(acc[r + 1], 0.f);
        *(v2h*)(hrow + r) = p;
      }
    }
  }
  __syncthreads();

  // ---- Phase 3: layer 2  out = relu(W2 * H + b2); one o2-tile per wave ----
  float* outb = out + b * kO2 * kN1;
  for (int ot = wave; ot < kO2 / 16; ot += 8) {      // exactly 1 o-tile per wave
    v16h Afr[kK2];                                    // 64 VGPRs of resident weights
    const _Float16* wrl = W2h + (ot * 16 + l15) * kO1 + hi8;
    #pragma unroll
    for (int k = 0; k < kK2; ++k) {
      v8h a0 = *(const v8h*)(wrl + 32 * k);
      v8h a1 = *(const v8h*)(wrl + 32 * k + 16);
      Afr[k] = __builtin_shufflevector(a0, a1, 0,1,2,3,4,5,6,7,8,9,10,11,12,13,14,15);
    }
    for (int nt = 0; nt < kNTT; ++nt) {
      const int nl = nt * 16 + l15;
      v8f acc;
      #pragma unroll
      for (int r = 0; r < 8; ++r) acc[r] = bias2[ot * 16 + hi8 + r];

      const _Float16* hrow = Hs + nl * kHR + hi * 8;
      #pragma unroll
      for (int k = 0; k < kK2; ++k) {
        v8h c0 = *(const v8h*)(hrow + 32 * k);
        v8h c1 = *(const v8h*)(hrow + 32 * k + 16);
        v16h Bm = __builtin_shufflevector(c0, c1, 0,1,2,3,4,5,6,7,8,9,10,11,12,13,14,15);
        acc = __builtin_amdgcn_wmma_f32_16x16x32_f16(false, Afr[k], false, Bm,
                                                     (short)0, acc, false, false);
      }
      const int ng = n0 + nt * 16 + l15;
      #pragma unroll
      for (int r = 0; r < 8; ++r) {                  // coalesced f32 stores per row
        const int o = ot * 16 + hi8 + r;
        outb[o * kN1 + ng] = fmaxf(acc[r], 0.f);
      }
    }
  }
}

// ---------------------------------------------------------------------------
extern "C" void kernel_launch(void* const* d_in, const int* in_sizes, int n_in,
                              void* d_out, int out_size, void* d_ws, size_t ws_size,
                              hipStream_t stream) {
  (void)in_sizes; (void)n_in; (void)out_size; (void)ws_size;
  const float* xyz1 = (const float*)d_in[0];
  const float* xyz2 = (const float*)d_in[1];
  const float* f1   = (const float*)d_in[2];
  const float* f2   = (const float*)d_in[3];
  const float* W1   = (const float*)d_in[4];
  const float* b1   = (const float*)d_in[5];
  const float* W2   = (const float*)d_in[6];
  const float* b2   = (const float*)d_in[7];
  float* out = (float*)d_out;

  // Workspace layout (all 16B-aligned offsets): 1.84 MB total
  char* ws = (char*)d_ws;
  _Float16* W1h = (_Float16*)ws;  ws += (size_t)kO1 * kCX * sizeof(_Float16);  // 196608
  _Float16* W2h = (_Float16*)ws;  ws += (size_t)kO2 * kO1 * sizeof(_Float16);  //  65536
  int*      idx3 = (int*)ws;      ws += (size_t)kB * kN1 * 3 * sizeof(int);    // 786432
  float*    w3   = (float*)ws;                                                 // 786432

  cvt_weights_kernel<<<(kO1 * kCX + kO2 * kO1) / 256, 256, 0, stream>>>(W1, W2, W1h, W2h);
  knn3_weights_kernel<<<dim3(kN1 / 256, kB), 256, 0, stream>>>(xyz1, xyz2, idx3, w3);
  fused_mlp_kernel<<<dim3(kN1 / kNT, kB), 256, kSmemBytes, stream>>>(
      f1, f2, b1, b2, W1h, W2h, idx3, w3, out);
}
